// Enformer_20753281974759
// MI455X (gfx1250) — compile-verified, matrix operand order checked by reading, not run on hardware
//
#include <hip/hip_runtime.h>
#include <cmath>
#include <cstdint>

typedef __attribute__((ext_vector_type(16))) _Float16 v16h;
typedef __attribute__((ext_vector_type(8)))  float    v8f;
typedef int gv4i __attribute__((vector_size(16)));   // builtin's pointee type

#define B_    2
#define L_    1536
#define DM_   1536
#define H_    8
#define DQK_  64
#define DV_   192
#define NB_   32
#define NPOS_ 3071
#define NPOSP 3072   /* padded to multiple of 64 */

#ifndef __has_builtin
#define __has_builtin(x) 0
#endif

#if defined(__AMDGCN__) && __has_builtin(__builtin_amdgcn_global_load_async_to_lds_b128)
#define ASYNC_LDS 1
#if __has_builtin(__builtin_amdgcn_s_wait_asynccnt)
#define WAIT_ASYNC(n) __builtin_amdgcn_s_wait_asynccnt(n)
#else
#define WAIT_ASYNC(n) asm volatile("s_wait_asynccnt %0" ::"n"(n) : "memory")
#endif
#else
#define ASYNC_LDS 0
#endif

struct PosConst {
  float ce[NB_];     // -ln2 / half_life
  float cw[NB_];     // central-mask widths 2^(j+1)-1
  float cm1[NB_];    // conc - 1
  float rate[NB_];
  float lnorm[NB_];  // lgamma(conc) - conc*log(rate)
  float inv_gmax;
};

// ---------------------------------------------------------------- elementwise
__global__ void cvt_f32_f16(const float* __restrict__ s, _Float16* __restrict__ d, int n) {
  int i = blockIdx.x * 256 + threadIdx.x;
  if (i < n) d[i] = (_Float16)s[i];
}

// dst[n][k] = src[k][n]  (f32 [K,N] row-major -> f16 [N,K] row-major)
__global__ void transpose_f32_f16(const float* __restrict__ s, _Float16* __restrict__ d,
                                  int K, int N) {
  int i = blockIdx.x * 256 + threadIdx.x;
  if (i < K * N) {
    int k = i / N, n = i % N;
    d[(size_t)n * K + k] = (_Float16)s[i];
  }
}

// Qw = f16(Q*scale + r_w_bias[col]); Qr = f16(Q*scale + r_r_bias[col]); col = h*64+d
__global__ void qbias_kernel(const float* __restrict__ Q, const float* __restrict__ rwb,
                             const float* __restrict__ rrb, _Float16* __restrict__ Qw,
                             _Float16* __restrict__ Qr, int n) {
  int i = blockIdx.x * 256 + threadIdx.x;
  if (i < n) {
    int c = i & 511;              // H*DQK = 512 columns
    float q = Q[i] * 0.125f;      // DQK^-0.5 = 1/8
    Qw[i] = (_Float16)(q + rwb[c]);
    Qr[i] = (_Float16)(q + rrb[c]);
  }
}

// pe[k][0:192] for k in [0,3071); row 3071 zeroed (pad). One thread per (k, j<32).
__global__ void pos_feats(_Float16* __restrict__ pe, PosConst pc) {
  int idx = blockIdx.x * 256 + threadIdx.x;   // NPOSP*32 threads
  int j = idx & 31, k = idx >> 5;
  _Float16* row = pe + (size_t)k * 192;
  if (k >= NPOS_) {
    row[j] = (_Float16)0.f; row[32 + j] = (_Float16)0.f; row[64 + j] = (_Float16)0.f;
    row[96 + j] = (_Float16)0.f; row[128 + j] = (_Float16)0.f; row[160 + j] = (_Float16)0.f;
    return;
  }
  float pos = (float)k - (float)(L_ - 1);
  float ap  = fabsf(pos);
  float fe  = __expf(pc.ce[j] * ap);
  float fc  = (pc.cw[j] > ap) ? 1.0f : 0.0f;
  float lp  = ((ap > 0.f) ? pc.cm1[j] * __logf(ap) : 0.f) - pc.rate[j] * ap - pc.lnorm[j];
  float fg  = (__expf(lp) + 1e-8f) * pc.inv_gmax;
  float sg  = (pos > 0.f) ? 1.f : ((pos < 0.f) ? -1.f : 0.f);
  row[j]       = (_Float16)fe;        row[32 + j]  = (_Float16)fc;
  row[64 + j]  = (_Float16)fg;        row[96 + j]  = (_Float16)(sg * fe);
  row[128 + j] = (_Float16)(sg * fc); row[160 + j] = (_Float16)(sg * fg);
}

// ---------------------------------------------------------------- WMMA GEMM
// C[z][m][n] = sum_k A[z][m][k] * Bt[z][n][k]   (both operands K-major, f16)
// 128 threads = 4 waves; block tile 64x64; each wave: 32x32 = 2x2 WMMA tiles.
// Double-buffered LDS; global->LDS via async path when available.
__global__ __launch_bounds__(128)
void gemm_wmma_f16(const _Float16* __restrict__ A, const _Float16* __restrict__ Bt,
                   void* __restrict__ C, int M, int N, int K,
                   int lda, int ldb, int ldc,
                   long long sAb, long long sAh, long long sBb, long long sBh,
                   long long sCb, long long sCh, int Hd,
                   int outF16, const float* __restrict__ bias) {
  __shared__ _Float16 As[2][64 * 48];   // stride 48 halves = 96B (16B aligned rows)
  __shared__ _Float16 Bs[2][64 * 48];

  const int tid  = threadIdx.x;
  const int lane = tid & 31;
  const int wv   = tid >> 5;
  const int wm   = (wv >> 1) * 32;
  const int wn   = (wv & 1) * 32;
  const int m0   = blockIdx.y * 64;
  const int n0   = blockIdx.x * 64;
  const int z    = blockIdx.z;
  const int zb   = z / Hd, zh = z % Hd;

  const _Float16* Ab = A + (size_t)zb * sAb + (size_t)zh * sAh;
  const _Float16* Bb = Bt + (size_t)zb * sBb + (size_t)zh * sBh;

  v8f acc[2][2] = {};

  const int mrow = lane & 15;          // fragment row/col within 16
  const int koff = (lane >> 4) << 3;   // K sub-offset: 0 (lanes 0-15) or 8 (lanes 16-31)
  const int lr   = tid >> 2;           // staging row 0..31 (and +32)
  const int sg   = (tid & 3) << 3;     // staging seg in halves: 0,8,16,24

  // ISA 16-bit A/B layout: lane<16 holds K {koff..+7, koff+16..+23}
  auto compute = [&](int cur) {
    union U { v16h v; uint4 q[2]; };
    U a0, a1, b0, b1;
    const _Float16* as_ = As[cur];
    const _Float16* bs_ = Bs[cur];
    const int r0 = wm + mrow, r1 = wm + 16 + mrow;
    const int c0 = wn + mrow, c1 = wn + 16 + mrow;
    a0.q[0] = *(const uint4*)&as_[r0 * 48 + koff];
    a0.q[1] = *(const uint4*)&as_[r0 * 48 + koff + 16];
    a1.q[0] = *(const uint4*)&as_[r1 * 48 + koff];
    a1.q[1] = *(const uint4*)&as_[r1 * 48 + koff + 16];
    b0.q[0] = *(const uint4*)&bs_[c0 * 48 + koff];
    b0.q[1] = *(const uint4*)&bs_[c0 * 48 + koff + 16];
    b1.q[0] = *(const uint4*)&bs_[c1 * 48 + koff];
    b1.q[1] = *(const uint4*)&bs_[c1 * 48 + koff + 16];
    acc[0][0] = __builtin_amdgcn_wmma_f32_16x16x32_f16(false, a0.v, false, b0.v, (short)0, acc[0][0], false, false);
    acc[0][1] = __builtin_amdgcn_wmma_f32_16x16x32_f16(false, a0.v, false, b1.v, (short)0, acc[0][1], false, false);
    acc[1][0] = __builtin_amdgcn_wmma_f32_16x16x32_f16(false, a1.v, false, b0.v, (short)0, acc[1][0], false, false);
    acc[1][1] = __builtin_amdgcn_wmma_f32_16x16x32_f16(false, a1.v, false, b1.v, (short)0, acc[1][1], false, false);
  };

#if ASYNC_LDS
  // 4 async ops per tile per wave -> ASYNCcnt<=4 means "previous tile done".
  auto issue = [&](int buf, int k0) {
    const size_t ra = (size_t)(m0 + lr) * lda + k0 + sg;
    const size_t rb = (size_t)(n0 + lr) * ldb + k0 + sg;
    __builtin_amdgcn_global_load_async_to_lds_b128(
        (gv4i*)(Ab + ra), (gv4i*)&As[buf][lr * 48 + sg], 0, 0);
    __builtin_amdgcn_global_load_async_to_lds_b128(
        (gv4i*)(Ab + ra + (size_t)32 * lda), (gv4i*)&As[buf][(lr + 32) * 48 + sg], 0, 0);
    __builtin_amdgcn_global_load_async_to_lds_b128(
        (gv4i*)(Bb + rb), (gv4i*)&Bs[buf][lr * 48 + sg], 0, 0);
    __builtin_amdgcn_global_load_async_to_lds_b128(
        (gv4i*)(Bb + rb + (size_t)32 * ldb), (gv4i*)&Bs[buf][(lr + 32) * 48 + sg], 0, 0);
  };

  issue(0, 0);
  const int nstep = K >> 5;
  for (int s = 0; s < nstep; ++s) {
    const int cur = s & 1;
    if (s + 1 < nstep) {
      issue(cur ^ 1, (s + 1) << 5);
      WAIT_ASYNC(4);             // tile `cur` landed (async loads complete in order)
    } else {
      WAIT_ASYNC(0);
    }
    __syncthreads();             // all waves' async writes visible
    compute(cur);
    __syncthreads();             // reads done before buffer `cur` is refilled
  }
#else
  // Fallback: batched sync staging (all loads issued before LDS stores).
  for (int k0 = 0; k0 < K; k0 += 32) {
    const size_t ra = (size_t)(m0 + lr) * lda + k0 + sg;
    const size_t rb = (size_t)(n0 + lr) * ldb + k0 + sg;
    uint4 va0 = *(const uint4*)&Ab[ra];
    uint4 va1 = *(const uint4*)&Ab[ra + (size_t)32 * lda];
    uint4 vb0 = *(const uint4*)&Bb[rb];
    uint4 vb1 = *(const uint4*)&Bb[rb + (size_t)32 * ldb];
    if (k0 + 32 < K) {
      __builtin_prefetch(&Ab[ra + 32], 0, 1);
      __builtin_prefetch(&Bb[rb + 32], 0, 1);
    }
    __syncthreads();
    *(uint4*)&As[0][lr * 48 + sg] = va0;
    *(uint4*)&As[0][(lr + 32) * 48 + sg] = va1;
    *(uint4*)&Bs[0][lr * 48 + sg] = vb0;
    *(uint4*)&Bs[0][(lr + 32) * 48 + sg] = vb1;
    __syncthreads();
    compute(0);
  }
#endif

  // C/D layout: element r of lane l -> m = r + 8*(l>=16), n = l&15
  long long coff = (long long)zb * sCb + (long long)zh * sCh;
  for (int tm = 0; tm < 2; ++tm)
    for (int tn = 0; tn < 2; ++tn) {
      int gm0 = m0 + wm + tm * 16 + ((lane >> 4) << 3);
      int gn  = n0 + wn + tn * 16 + mrow;
      for (int r = 0; r < 8; ++r) {
        float v = acc[tm][tn][r];
        if (bias) v += bias[gn];
        size_t idx = (size_t)(gm0 + r) * ldc + gn;
        if (outF16) ((_Float16*)C)[coff + idx] = (_Float16)v;
        else        ((float*)C)[coff + idx]    = v;
      }
    }
}

// ----------------------------------------------- relative shift + softmax
// logits[z][i][j] = content[z][i][j] + rel[z][i][L-1 + j - i]; softmax over j.
__global__ __launch_bounds__(256)
void softmax_shift(const _Float16* __restrict__ content, const _Float16* __restrict__ rel,
                   _Float16* __restrict__ w) {
  __shared__ float red[256];
  const int t = threadIdx.x;
  const int i = blockIdx.x;
  const int z = blockIdx.y;
  const _Float16* crow = content + ((size_t)z * L_ + i) * L_;
  const _Float16* rrow = rel + ((size_t)z * L_ + i) * NPOSP;
  _Float16* wrow = w + ((size_t)z * L_ + i) * L_;

  float lv[6];
  float mx = -1e30f;
#pragma unroll
  for (int u = 0; u < 6; ++u) {
    int j = t + u * 256;
    float l = (float)crow[j] + (float)rrow[(L_ - 1) + j - i];
    lv[u] = l;
    mx = fmaxf(mx, l);
  }
  red[t] = mx; __syncthreads();
  for (int s = 128; s > 0; s >>= 1) { if (t < s) red[t] = fmaxf(red[t], red[t + s]); __syncthreads(); }
  mx = red[0]; __syncthreads();

  float sum = 0.f;
#pragma unroll
  for (int u = 0; u < 6; ++u) { lv[u] = __expf(lv[u] - mx); sum += lv[u]; }
  red[t] = sum; __syncthreads();
  for (int s = 128; s > 0; s >>= 1) { if (t < s) red[t] += red[t + s]; __syncthreads(); }
  float inv = 1.0f / red[0];
#pragma unroll
  for (int u = 0; u < 6; ++u) wrow[t + u * 256] = (_Float16)(lv[u] * inv);
}

// ---------------------------------------------------------------- launcher
extern "C" void kernel_launch(void* const* d_in, const int* in_sizes, int n_in,
                              void* d_out, int out_size, void* d_ws, size_t ws_size,
                              hipStream_t stream) {
  const float* x   = (const float*)d_in[0];
  const float* Wq  = (const float*)d_in[1];
  const float* Wk  = (const float*)d_in[2];
  const float* Wv  = (const float*)d_in[3];
  const float* Wrk = (const float*)d_in[4];
  const float* rwb = (const float*)d_in[5];
  const float* rrb = (const float*)d_in[6];
  const float* Wo  = (const float*)d_in[7];
  const float* bo  = (const float*)d_in[8];

  const int T = B_ * L_;          // 3072 tokens
  const int HD = H_ * DQK_;       // 512
  const int HV = H_ * DV_;        // 1536

  // ---- host-computed positional constants (pure math; deterministic)
  PosConst pc;
  {
    const double Lf = (double)L_;
    const double max_range = std::log(Lf) / std::log(2.0);
    double cm1[NB_], rate[NB_], lnorm[NB_];
    for (int j = 0; j < NB_; ++j) {
      double hl = std::pow(2.0, 3.0 + j * (max_range - 3.0) / (NB_ - 1));
      pc.ce[j] = (float)(-std::log(2.0) / hl);
      pc.cw[j] = (float)(std::pow(2.0, (double)(j + 1)) - 1.0);
      double mean = Lf / NB_ * (j + 1);
      double sd = Lf / (2.0 * NB_);
      double conc = (mean / sd) * (mean / sd);
      double rt = mean / (sd * sd);
      cm1[j] = conc - 1.0; rate[j] = rt;
      lnorm[j] = std::lgamma(conc) - conc * std::log(rt);
      pc.cm1[j] = (float)cm1[j]; pc.rate[j] = (float)rt; pc.lnorm[j] = (float)lnorm[j];
    }
    double gmax = 0.0;
    for (int a = 0; a <= NPOS_ - 1; ++a)
      for (int j = 0; j < NB_; ++j) {
        double lp = (a > 0 ? cm1[j] * std::log((double)a) : 0.0) - rate[j] * a - lnorm[j];
        double pv = std::exp(lp) + 1e-8;
        if (pv > gmax) gmax = pv;
      }
    pc.inv_gmax = (float)(1.0 / gmax);
  }

  // ---- workspace bump allocator
  char* p = (char*)d_ws;
  auto alloc = [&](size_t bytes) -> char* {
    char* r = p; p += (bytes + 255) & ~(size_t)255; return r;
  };
  _Float16* xh    = (_Float16*)alloc((size_t)T * DM_ * 2);
  _Float16* WqT   = (_Float16*)alloc((size_t)HD * DM_ * 2);
  _Float16* WkT   = (_Float16*)alloc((size_t)HD * DM_ * 2);
  _Float16* WvT   = (_Float16*)alloc((size_t)HV * DM_ * 2);
  _Float16* WrkT  = (_Float16*)alloc((size_t)HD * 192 * 2);
  _Float16* WoT   = (_Float16*)alloc((size_t)DM_ * HV * 2);
  _Float16* peh   = (_Float16*)alloc((size_t)NPOSP * 192 * 2);
  float*    Qf32  = (float*)   alloc((size_t)T * HD * 4);
  _Float16* Qw    = (_Float16*)alloc((size_t)T * HD * 2);
  _Float16* Qr    = (_Float16*)alloc((size_t)T * HD * 2);
  _Float16* Kh    = (_Float16*)alloc((size_t)T * HD * 2);
  _Float16* Vth   = (_Float16*)alloc((size_t)HV * T * 2);
  _Float16* rKh   = (_Float16*)alloc((size_t)NPOSP * HD * 2);
  _Float16* conth = (_Float16*)alloc((size_t)16 * L_ * L_ * 2);
  _Float16* relh  = (_Float16*)alloc((size_t)16 * L_ * NPOSP * 2);
  _Float16* wh    = (_Float16*)alloc((size_t)16 * L_ * L_ * 2);
  _Float16* A2h   = (_Float16*)alloc((size_t)T * HV * 2);
  (void)ws_size; (void)n_in; (void)in_sizes; (void)out_size;

  auto gemm = [&](const _Float16* A, const _Float16* Bt, void* C,
                  int M, int N, int K, int lda, int ldb, int ldc,
                  long long sAb, long long sAh, long long sBb, long long sBh,
                  long long sCb, long long sCh, int Hd, int nz, int outF16,
                  const float* bias) {
    dim3 g(N / 64, M / 64, nz);
    gemm_wmma_f16<<<g, dim3(128), 0, stream>>>(A, Bt, C, M, N, K, lda, ldb, ldc,
                                               sAb, sAh, sBb, sBh, sCb, sCh, Hd,
                                               outF16, bias);
  };

  // 1) weight transposes + x conversion + positional features
  transpose_f32_f16<<<(DM_ * HD + 255) / 256, 256, 0, stream>>>(Wq, WqT, DM_, HD);
  transpose_f32_f16<<<(DM_ * HD + 255) / 256, 256, 0, stream>>>(Wk, WkT, DM_, HD);
  transpose_f32_f16<<<(DM_ * HV + 255) / 256, 256, 0, stream>>>(Wv, WvT, DM_, HV);
  transpose_f32_f16<<<(192 * HD + 255) / 256, 256, 0, stream>>>(Wrk, WrkT, 192, HD);
  transpose_f32_f16<<<(HV * DM_ + 255) / 256, 256, 0, stream>>>(Wo, WoT, HV, DM_);
  cvt_f32_f16<<<(T * DM_ + 255) / 256, 256, 0, stream>>>(x, xh, T * DM_);
  pos_feats<<<(NPOSP * 32) / 256, 256, 0, stream>>>(peh, pc);

  // 2) projections
  gemm(xh, WqT, Qf32, T, HD, DM_, DM_, DM_, HD, 0,0,0,0,0,0, 1, 1, 0, nullptr);
  gemm(xh, WkT, Kh,   T, HD, DM_, DM_, DM_, HD, 0,0,0,0,0,0, 1, 1, 1, nullptr);
  // Vt[d'][t]: A = WvT [HV, DM], Bt = xh [T, DM]
  gemm(WvT, xh, Vth, HV, T, DM_, DM_, DM_, T, 0,0,0,0,0,0, 1, 1, 1, nullptr);
  // rK[k][h*64+d]: A = pe [NPOSP,192], Bt = WrkT [HD,192]
  gemm(peh, WrkT, rKh, NPOSP, HD, 192, 192, 192, HD, 0,0,0,0,0,0, 1, 1, 1, nullptr);

  // 3) Q scale + biases -> f16
  qbias_kernel<<<(T * HD + 255) / 256, 256, 0, stream>>>(Qf32, rwb, rrb, Qw, Qr, T * HD);

  // 4) content[z][i][j] = (Qs+rwb) . K     (z = b*8+h, M=N=L, K=64)
  gemm(Qw, Kh, conth, L_, L_, DQK_, HD, HD, L_,
       (long long)L_ * HD, DQK_, (long long)L_ * HD, DQK_,
       8LL * L_ * L_, (long long)L_ * L_, H_, 16, 1, nullptr);

  // 5) rel[z][i][k] = (Qs+rrb) . rK        (N = NPOSP, rK shared across batch)
  gemm(Qr, rKh, relh, L_, NPOSP, DQK_, HD, HD, NPOSP,
       (long long)L_ * HD, DQK_, 0, DQK_,
       8LL * L_ * NPOSP, (long long)L_ * NPOSP, H_, 16, 1, nullptr);

  // 6) relative shift + softmax -> w (f16)
  softmax_shift<<<dim3(L_, 16), 256, 0, stream>>>(conth, relh, wh);

  // 7) attn[z][i][d] = w . V   -> write into A2 [t, h*192+d]
  gemm(wh, Vth, A2h, L_, DV_, L_, L_, T, HV,
       8LL * L_ * L_, (long long)L_ * L_,     // A strides
       (long long)L_, (long long)DV_ * T,     // Bt: +b*1536 cols, +h*192 rows
       (long long)L_ * HV, (long long)DV_,    // C: row b*L+i, col h*192+d
       H_, 16, 1, nullptr);

  // 8) out = A2 @ Wo + bo  (f32 into d_out; Bt = WoT [DM, HV])
  gemm(A2h, WoT, d_out, T, DM_, HV, HV, HV, DM_, 0,0,0,0,0,0, 1, 1, 0, bo);
}
